// SimpleModel_8349416423650
// MI455X (gfx1250) — matrix operand order repaired and checked by reading, not compiled, not run on hardware
//
#include <hip/hip_runtime.h>
#include <hip/hip_bf16.h>

// GIN 2-layer forward for MI455X (gfx1250, wave32).
// Bottleneck: edge scatter/gather through L2 (~4.4 GB on-chip traffic);
// MLPs are tiny (~2.3 GFLOP) and run on v_wmma_f32_16x16x32_f16.

#define N_NODES  200000
#define N_EDGES  6400000
#define N_GRAPHS 256
#define BN_EPS   1e-5f

typedef __attribute__((ext_vector_type(16))) _Float16 v16h;
typedef __attribute__((ext_vector_type(8)))  float    v8f;

__device__ __forceinline__ v8f wmma16(v16h a, v16h b, v8f c) {
  // D = A(16x32 f16) x B(32x16 f16) + C(16x16 f32)
  return __builtin_amdgcn_wmma_f32_16x16x32_f16(
      /*neg_a=*/false, a, /*neg_b=*/false, b,
      /*c_mod=*/(short)0, c, /*reuse_a=*/false, /*reuse_b=*/false);
}

// ---------------------------------------------------------------- utilities
__global__ void k_zero(float* __restrict__ p, int n) {
  int i = blockIdx.x * blockDim.x + threadIdx.x;
  if (i < n) p[i] = 0.0f;
}

__global__ void k_copy4(const float4* __restrict__ s, float4* __restrict__ d, int n4) {
  int i = blockIdx.x * blockDim.x + threadIdx.x;
  if (i < n4) d[i] = s[i];
}

// Pack weight W[Kreal x Ncols] (f32, row-major) into WMMA B-operand order,
// zero-padded to K = KB*32, N = NT*16.  Element t maps to
// out[((nt*KB+kb)*32 + lane)*16 + e] with K = kb*32 + 16*(e>>3) + 8*(lane>>4) + (e&7),
// col = nt*16 + (lane&15)   (ISA 7.12.2 B layout, 16-bit, wave32).
__global__ void k_pack(const float* __restrict__ W, _Float16* __restrict__ out,
                       int Kreal, int Ncols, int KB, int NT) {
  int t = blockIdx.x * blockDim.x + threadIdx.x;
  int total = KB * NT * 512;
  if (t >= total) return;
  int e    = t & 15;
  int lane = (t >> 4) & 31;
  int blk  = t >> 9;
  int kb   = blk % KB;
  int nt   = blk / KB;
  int k    = kb * 32 + ((e >= 8) ? 16 : 0) + 8 * (lane >> 4) + (e & 7);
  int col  = nt * 16 + (lane & 15);
  float v  = (k < Kreal && col < Ncols) ? W[k * Ncols + col] : 0.0f;
  out[t] = (_Float16)v;
}

// ------------------------------------------------------- edge scatter (L2)
__global__ void k_edge16(const int* __restrict__ ei, const float* __restrict__ x,
                         float* __restrict__ agg) {
  int e = blockIdx.x * blockDim.x + threadIdx.x;
  if (e >= N_EDGES) return;
  __builtin_prefetch(ei + e + 8192, 0, 1);            // global_prefetch_b8
  __builtin_prefetch(ei + N_EDGES + e + 8192, 0, 1);
  int src = ei[e];
  int dst = ei[N_EDGES + e];
  const float4* s = (const float4*)(x + (size_t)src * 16);
  float* d = agg + (size_t)dst * 16;
  float4 v0 = s[0], v1 = s[1], v2 = s[2], v3 = s[3];
  atomicAdd(d +  0, v0.x); atomicAdd(d +  1, v0.y); atomicAdd(d +  2, v0.z); atomicAdd(d +  3, v0.w);
  atomicAdd(d +  4, v1.x); atomicAdd(d +  5, v1.y); atomicAdd(d +  6, v1.z); atomicAdd(d +  7, v1.w);
  atomicAdd(d +  8, v2.x); atomicAdd(d +  9, v2.y); atomicAdd(d + 10, v2.z); atomicAdd(d + 11, v2.w);
  atomicAdd(d + 12, v3.x); atomicAdd(d + 13, v3.y); atomicAdd(d + 14, v3.z); atomicAdd(d + 15, v3.w);
}

// 4 threads per edge, each moves a 16-float chunk of the 64-feature row.
__global__ void k_edge64(const int* __restrict__ ei, const float* __restrict__ h,
                         float* __restrict__ agg) {
  int t = blockIdx.x * blockDim.x + threadIdx.x;
  if (t >= N_EDGES * 4) return;
  int e = t >> 2, c = t & 3;
  __builtin_prefetch(ei + e + 8192, 0, 1);
  int src = ei[e];
  int dst = ei[N_EDGES + e];
  const float4* s = (const float4*)(h + (size_t)src * 64 + c * 16);
  float* d = agg + (size_t)dst * 64 + c * 16;
  float4 v0 = s[0], v1 = s[1], v2 = s[2], v3 = s[3];
  atomicAdd(d +  0, v0.x); atomicAdd(d +  1, v0.y); atomicAdd(d +  2, v0.z); atomicAdd(d +  3, v0.w);
  atomicAdd(d +  4, v1.x); atomicAdd(d +  5, v1.y); atomicAdd(d +  6, v1.z); atomicAdd(d +  7, v1.w);
  atomicAdd(d +  8, v2.x); atomicAdd(d +  9, v2.y); atomicAdd(d + 10, v2.z); atomicAdd(d + 11, v2.w);
  atomicAdd(d + 12, v3.x); atomicAdd(d + 13, v3.y); atomicAdd(d + 14, v3.z); atomicAdd(d + 15, v3.w);
}

// ------------------------------------------------------------- MLP layer 1
// per wave: one 16-node tile; 16->64 (K padded to 32) then 64->64, WMMA f16.
// Fused BN-statistics reduction (sum, sum^2 per feature) via LDS atomics.
__global__ void __launch_bounds__(128) k_mlp1(
    const float* __restrict__ agg0,
    const _Float16* __restrict__ pW1a, const _Float16* __restrict__ pW1b,
    const float* __restrict__ b1a, const float* __restrict__ b1b,
    float* __restrict__ z1, float* __restrict__ stats1) {
  __shared__ _Float16 mid[4][16][64];
  __shared__ float ssum[64], ssq[64];
  const int lane = threadIdx.x & 31;
  const int wave = threadIdx.x >> 5;
  const int tile = blockIdx.x * 4 + wave;      // 3125 blocks * 4 = 12500 tiles
  const int row  = lane & 15;                  // A row == C column == lane&15
  const int half = lane >> 4;
  if (threadIdx.x < 64) { ssum[threadIdx.x] = 0.0f; ssq[threadIdx.x] = 0.0f; }
  __syncthreads();

  // ---- stage 1: A = agg0 tile (16x16, K zero-padded to 32) ----
  const float4* rp = (const float4*)(agg0 + (size_t)(tile * 16 + row) * 16);
  float4 u = rp[2 * half + 0];
  float4 v = rp[2 * half + 1];
  v16h a0 = {};
  a0[0] = (_Float16)u.x; a0[1] = (_Float16)u.y; a0[2] = (_Float16)u.z; a0[3] = (_Float16)u.w;
  a0[4] = (_Float16)v.x; a0[5] = (_Float16)v.y; a0[6] = (_Float16)v.z; a0[7] = (_Float16)v.w;
  // e=8..15 correspond to K>=16 -> zero (input feature dim is 16)

#pragma unroll
  for (int nt = 0; nt < 4; ++nt) {
    v16h b = *(const v16h*)(pW1a + (nt * 32 + lane) * 16);
    v8f c = {};
    c = wmma16(a0, b, c);
    float bias = b1a[nt * 16 + row];
#pragma unroll
    for (int r = 0; r < 8; ++r) {
      float vv = fmaxf(c[r] + bias, 0.0f);           // relu
      mid[wave][r + 8 * half][nt * 16 + row] = (_Float16)vv;
    }
  }
  __syncthreads();

  // ---- stage 2: mid(16x64) x W1b(64x64) ----
#pragma unroll
  for (int nt = 0; nt < 4; ++nt) {
    v8f c = {};
#pragma unroll
    for (int kb = 0; kb < 2; ++kb) {
      v16h a;
#pragma unroll
      for (int e = 0; e < 16; ++e) {
        int k = kb * 32 + ((e >= 8) ? 16 : 0) + 8 * half + (e & 7);
        a[e] = mid[wave][row][k];
      }
      v16h b = *(const v16h*)(pW1b + ((nt * 2 + kb) * 32 + lane) * 16);
      c = wmma16(a, b, c);
    }
    float bias = b1b[nt * 16 + row];
    float s = 0.0f, q = 0.0f;
#pragma unroll
    for (int r = 0; r < 8; ++r) {
      float vv = c[r] + bias;
      z1[(size_t)(tile * 16 + r + 8 * half) * 64 + nt * 16 + row] = vv;
      s += vv; q += vv * vv;
    }
    atomicAdd(&ssum[nt * 16 + row], s);
    atomicAdd(&ssq [nt * 16 + row], q);
  }
  __syncthreads();
  if (threadIdx.x < 64) {
    atomicAdd(&stats1[threadIdx.x],      ssum[threadIdx.x]);
    atomicAdd(&stats1[64 + threadIdx.x], ssq [threadIdx.x]);
  }
}

// ------------------------------------------------------------- MLP layer 2
// 64->8 (N padded to 16) then 8->8 (K,N padded), fused BN stats.
__global__ void __launch_bounds__(128) k_mlp2(
    const float* __restrict__ agg1,
    const _Float16* __restrict__ pW2a, const _Float16* __restrict__ pW2b,
    const float* __restrict__ b2a, const float* __restrict__ b2b,
    float* __restrict__ z2, float* __restrict__ stats2) {
  __shared__ _Float16 mid[4][16][16];
  __shared__ float ssum[8], ssq[8];
  const int lane = threadIdx.x & 31;
  const int wave = threadIdx.x >> 5;
  const int tile = blockIdx.x * 4 + wave;
  const int row  = lane & 15;
  const int half = lane >> 4;
  if (threadIdx.x < 8) { ssum[threadIdx.x] = 0.0f; ssq[threadIdx.x] = 0.0f; }
  __syncthreads();

  const float4* rp = (const float4*)(agg1 + (size_t)(tile * 16 + row) * 64);
  v8f c = {};
#pragma unroll
  for (int kb = 0; kb < 2; ++kb) {
    float4 p0 = rp[kb * 8 + 2 * half + 0];
    float4 p1 = rp[kb * 8 + 2 * half + 1];
    float4 p2 = rp[kb * 8 + 4 + 2 * half + 0];
    float4 p3 = rp[kb * 8 + 4 + 2 * half + 1];
    v16h a;
    a[ 0] = (_Float16)p0.x; a[ 1] = (_Float16)p0.y; a[ 2] = (_Float16)p0.z; a[ 3] = (_Float16)p0.w;
    a[ 4] = (_Float16)p1.x; a[ 5] = (_Float16)p1.y; a[ 6] = (_Float16)p1.z; a[ 7] = (_Float16)p1.w;
    a[ 8] = (_Float16)p2.x; a[ 9] = (_Float16)p2.y; a[10] = (_Float16)p2.z; a[11] = (_Float16)p2.w;
    a[12] = (_Float16)p3.x; a[13] = (_Float16)p3.y; a[14] = (_Float16)p3.z; a[15] = (_Float16)p3.w;
    v16h b = *(const v16h*)(pW2a + (kb * 32 + lane) * 16);
    c = wmma16(a, b, c);
  }
  float bias = (row < 8) ? b2a[row] : 0.0f;      // cols 8..15 are zero pad
#pragma unroll
  for (int r = 0; r < 8; ++r) {
    float vv = fmaxf(c[r] + bias, 0.0f);
    mid[wave][r + 8 * half][row] = (_Float16)vv;
  }
  __syncthreads();

  v16h a2 = {};
#pragma unroll
  for (int e = 0; e < 8; ++e)                    // K real = 8 (rest zero)
    a2[e] = mid[wave][row][8 * half + e];
  v16h b2 = *(const v16h*)(pW2b + lane * 16);
  v8f c2 = {};
  c2 = wmma16(a2, b2, c2);
  if (row < 8) {
    float bias2 = b2b[row];
    float s = 0.0f, q = 0.0f;
#pragma unroll
    for (int r = 0; r < 8; ++r) {
      float vv = c2[r] + bias2;
      z2[(size_t)(tile * 16 + r + 8 * half) * 8 + row] = vv;
      s += vv; q += vv * vv;
    }
    atomicAdd(&ssum[row], s);
    atomicAdd(&ssq [row], q);
  }
  __syncthreads();
  if (threadIdx.x < 8) {
    atomicAdd(&stats2[threadIdx.x],     ssum[threadIdx.x]);
    atomicAdd(&stats2[8 + threadIdx.x], ssq [threadIdx.x]);
  }
}

// --------------------------------------------------------------- BN + pool
__global__ void k_bn_fin(const float* __restrict__ stats,
                         const float* __restrict__ gamma, const float* __restrict__ beta,
                         float* __restrict__ scale, float* __restrict__ shift,
                         int F, float invN) {
  int f = threadIdx.x;
  if (f >= F) return;
  float mean = stats[f] * invN;
  float var  = stats[F + f] * invN - mean * mean;
  float inv  = rsqrtf(var + BN_EPS);
  float sc   = gamma[f] * inv;
  scale[f] = sc;
  shift[f] = beta[f] - mean * sc;
}

__global__ void k_bn_relu(float* __restrict__ z, const float* __restrict__ scale,
                          const float* __restrict__ shift, int total, int fmask) {
  int i = blockIdx.x * blockDim.x + threadIdx.x;
  if (i >= total) return;
  int f = i & fmask;
  z[i] = fmaxf(z[i] * scale[f] + shift[f], 0.0f);
}

__global__ void k_pool(const float* __restrict__ h2, const int* __restrict__ batch,
                       float* __restrict__ pooled, float* __restrict__ counts) {
  int n = blockIdx.x * blockDim.x + threadIdx.x;
  if (n >= N_NODES) return;
  int b = batch[n];
  const float4* r = (const float4*)(h2 + (size_t)n * 8);
  float4 a = r[0], c = r[1];
  float* p = pooled + b * 8;
  atomicAdd(p + 0, a.x); atomicAdd(p + 1, a.y); atomicAdd(p + 2, a.z); atomicAdd(p + 3, a.w);
  atomicAdd(p + 4, c.x); atomicAdd(p + 5, c.y); atomicAdd(p + 6, c.z); atomicAdd(p + 7, c.w);
  atomicAdd(counts + b, 1.0f);
}

__global__ void k_out(const float* __restrict__ pooled, const float* __restrict__ counts,
                      float* __restrict__ out) {
  int i = blockIdx.x * blockDim.x + threadIdx.x;
  if (i >= N_GRAPHS * 8) return;
  out[i] = pooled[i] / fmaxf(counts[i >> 3], 1.0f);
}

// ----------------------------------------------------------------- driver
extern "C" void kernel_launch(void* const* d_in, const int* in_sizes, int n_in,
                              void* d_out, int out_size, void* d_ws, size_t ws_size,
                              hipStream_t stream) {
  const float* x   = (const float*)d_in[0];
  const int*   ei  = (const int*)d_in[1];
  const int*   bat = (const int*)d_in[2];
  const float* W1a = (const float*)d_in[3];  const float* b1a = (const float*)d_in[4];
  const float* W1b = (const float*)d_in[5];  const float* b1b = (const float*)d_in[6];
  const float* W2a = (const float*)d_in[7];  const float* b2a = (const float*)d_in[8];
  const float* W2b = (const float*)d_in[9];  const float* b2b = (const float*)d_in[10];
  const float* g1  = (const float*)d_in[11]; const float* be1 = (const float*)d_in[12];
  const float* g2  = (const float*)d_in[13]; const float* be2 = (const float*)d_in[14];

  // workspace layout (needs ~122 MB)
  char* ws = (char*)d_ws;
  float* agg0 = (float*)(ws);                  // 200000*16 f32 = 12.8 MB
  float* z1   = (float*)(ws +  12800000);      // 200000*64 f32 = 51.2 MB (z1 -> h1 in place)
  float* agg1 = (float*)(ws +  64000000);      // 200000*64 f32 = 51.2 MB
  float* z2   = (float*)(ws + 115200000);      // 200000*8  f32 =  6.4 MB (z2 -> h2 in place)
  float* smf  = (float*)(ws + 121600000);      // small accumulators
  float* stats1 = smf + 0;     // 128 (sum,sumsq)
  float* stats2 = smf + 128;   // 16
  float* sc1    = smf + 144;   // 64
  float* sh1    = smf + 208;   // 64
  float* sc2    = smf + 272;   // 8
  float* sh2    = smf + 280;   // 8
  float* pooled = smf + 288;   // 2048
  float* counts = smf + 2336;  // 256
  _Float16* wpack = (_Float16*)(ws + 121632768);
  _Float16* pW1a = wpack + 0;     // 2048 halves (K 16->32, N 64)
  _Float16* pW1b = wpack + 2048;  // 4096 halves (K 64,     N 64)
  _Float16* pW2a = wpack + 6144;  // 1024 halves (K 64,     N 8->16)
  _Float16* pW2b = wpack + 7168;  //  512 halves (K 8->32,  N 8->16)

  const int B = 256;
  // re-initialize accumulators every call (graph-replay safe)
  k_zero<<<(2592 + B - 1) / B, B, 0, stream>>>(smf, 2592);
  k_pack<<<(2048 + B - 1) / B, B, 0, stream>>>(W1a, pW1a, 16, 64, 1, 4);
  k_pack<<<(4096 + B - 1) / B, B, 0, stream>>>(W1b, pW1b, 64, 64, 2, 4);
  k_pack<<<(1024 + B - 1) / B, B, 0, stream>>>(W2a, pW2a, 64,  8, 2, 1);
  k_pack<<<( 512 + B - 1) / B, B, 0, stream>>>(W2b, pW2b,  8,  8, 1, 1);

  // layer 1: agg0 = x + segment_sum(x[src] -> dst)
  k_copy4<<<(800000 + B - 1) / B, B, 0, stream>>>((const float4*)x, (float4*)agg0, 800000);
  k_edge16<<<(N_EDGES + B - 1) / B, B, 0, stream>>>(ei, x, agg0);
  k_mlp1<<<3125, 128, 0, stream>>>(agg0, pW1a, pW1b, b1a, b1b, z1, stats1);
  k_bn_fin<<<1, 64, 0, stream>>>(stats1, g1, be1, sc1, sh1, 64, 1.0f / N_NODES);
  k_bn_relu<<<(12800000 + B - 1) / B, B, 0, stream>>>(z1, sc1, sh1, 12800000, 63);

  // layer 2: agg1 = h1 + segment_sum(h1[src] -> dst)
  k_copy4<<<(3200000 + B - 1) / B, B, 0, stream>>>((const float4*)z1, (float4*)agg1, 3200000);
  k_edge64<<<(N_EDGES * 4 + B - 1) / B, B, 0, stream>>>(ei, z1, agg1);
  k_mlp2<<<3125, 128, 0, stream>>>(agg1, pW2a, pW2b, b2a, b2b, z2, stats2);
  k_bn_fin<<<1, 64, 0, stream>>>(stats2, g2, be2, sc2, sh2, 8, 1.0f / N_NODES);
  k_bn_relu<<<(1600000 + B - 1) / B, B, 0, stream>>>(z2, sc2, sh2, 1600000, 7);

  // scatter-mean pooling
  k_pool<<<(N_NODES + B - 1) / B, B, 0, stream>>>(z2, bat, pooled, counts);
  k_out<<<(2048 + B - 1) / B, B, 0, stream>>>(pooled, counts, (float*)d_out);
}
